// HighOrderTransformerBlock_67061619360470
// MI455X (gfx1250) — matrix operand order repaired and checked by reading, not compiled, not run on hardware
//
#include <hip/hip_runtime.h>
#include <cstdint>
#include <cstddef>

// ---------------- problem dims ----------------
constexpr int B_ = 4, T_ = 256, N_ = 64, D_ = 512, H_ = 8, DF_ = 2048;
constexpr int M_ = B_ * T_ * N_;   // 65536 token rows
constexpr float EPS_ = 1e-6f;

// ---------------- vector types ----------------
typedef __bf16        v16bf __attribute__((ext_vector_type(16)));
typedef float         v8f   __attribute__((ext_vector_type(8)));
typedef float         v2f   __attribute__((ext_vector_type(2)));
typedef unsigned int  v4u   __attribute__((ext_vector_type(4)));
typedef unsigned int  u32x4 __attribute__((ext_vector_type(4)));
typedef int           i32x8 __attribute__((ext_vector_type(8)));
typedef int           i32x4 __attribute__((ext_vector_type(4)));

union FragU { struct { v4u lo, hi; } s; v16bf bf; };

__device__ __forceinline__ unsigned short f2bf(float f) {
  unsigned u = __float_as_uint(f);
  u += 0x7FFFu + ((u >> 16) & 1u);          // round-to-nearest-even
  return (unsigned short)(u >> 16);
}
__device__ __forceinline__ float bf2f(unsigned short s) {
  return __uint_as_float(((unsigned)s) << 16);
}

#if __has_builtin(__builtin_amdgcn_tensor_load_to_lds) && \
    __has_builtin(__builtin_amdgcn_s_wait_tensorcnt)
#define USE_TDM 1
#else
#define USE_TDM 0
#endif

#if __has_builtin(__builtin_amdgcn_wmma_f32_16x16x4_f32)
#define USE_WMMA_F32 1
#else
#define USE_WMMA_F32 0
#endif

#if USE_TDM
// Issue one TDM 2D tile load: `rows` rows of `dw_row` dwords, global row
// stride `dw_stride` dwords, into LDS at byte offset lds_addr.
__device__ __forceinline__ void tdm_load_tile(unsigned lds_addr, const void* gptr,
                                              unsigned rows, unsigned dw_row,
                                              unsigned dw_stride) {
  unsigned long long ga = (unsigned long long)gptr;
  u32x4 g0 = { 1u, lds_addr, (unsigned)ga,
               (unsigned)(ga >> 32) | (2u << 30) };          // count=1, type=2
  const unsigned td0 = dw_stride;
  const unsigned td1 = 1u << 20;
  i32x8 g1 = { (int)(2u << 16),                               // data_size = 4B
               (int)((td0 & 0xFFFFu) << 16),
               (int)((td0 >> 16) | ((td1 & 0xFFFFu) << 16)),
               (int)((td1 >> 16) | (dw_row << 16)),
               (int)rows,                                     // tile_dim1
               (int)dw_stride,                                // dim0 stride
               0, 0 };
  i32x4 z = { 0, 0, 0, 0 };
#if defined(__clang_major__) && (__clang_major__ >= 23)
  i32x8 z8 = { 0, 0, 0, 0, 0, 0, 0, 0 };
  __builtin_amdgcn_tensor_load_to_lds(g0, g1, z, z, z8, 0);
#else
  __builtin_amdgcn_tensor_load_to_lds(g0, g1, z, z, 0);
#endif
}
#endif

// ============================================================
// Weight transpose + f32->bf16:  Win[K][Ncol] -> Wout[Ncol][K]
// ============================================================
__global__ void wT_bf16_kernel(const float* __restrict__ Win,
                               unsigned short* __restrict__ Wout,
                               int K, int Ncol) {
  int idx = blockIdx.x * blockDim.x + threadIdx.x;
  if (idx >= K * Ncol) return;
  int n = idx / K, k = idx % K;
  Wout[idx] = f2bf(Win[(size_t)k * Ncol + n]);
}

// ============================================================
// RMSNorm (over D=512) -> bf16.  One wave per row, 8 rows/block.
// ============================================================
__global__ void __launch_bounds__(256)
rmsnorm_bf16_kernel(const float* __restrict__ X, const float* __restrict__ W,
                    unsigned short* __restrict__ Y, int Mrows) {
  int wave = threadIdx.x >> 5;
  int lane = threadIdx.x & 31;
  int row  = blockIdx.x * 8 + wave;
  if (row >= Mrows) return;
  const float4* Xr = (const float4*)(X + (size_t)row * D_);
  float4 xv[4];
  float ss = 0.f;
#pragma unroll
  for (int e = 0; e < 4; ++e) {
    xv[e] = Xr[e * 32 + lane];
    ss += xv[e].x * xv[e].x + xv[e].y * xv[e].y +
          xv[e].z * xv[e].z + xv[e].w * xv[e].w;
  }
#pragma unroll
  for (int off = 16; off; off >>= 1) ss += __shfl_xor(ss, off, 32);
  float sc = rsqrtf(ss * (1.0f / (float)D_) + EPS_);
  const float4* Wr = (const float4*)W;
#pragma unroll
  for (int e = 0; e < 4; ++e) {
    float4 wv = Wr[e * 32 + lane];
    ushort4 o;
    o.x = f2bf(xv[e].x * sc * wv.x);
    o.y = f2bf(xv[e].y * sc * wv.y);
    o.z = f2bf(xv[e].z * sc * wv.z);
    o.w = f2bf(xv[e].w * sc * wv.w);
    *(ushort4*)(Y + (size_t)row * D_ + (size_t)(e * 32 + lane) * 4) = o;
  }
}

// ============================================================
// bf16 WMMA GEMM (NT):  C[M,N] = A[M,K](bf16) * BT[N,K](bf16)^T
// 128x128 block tile, BK=64, TDM double-buffered LDS staging.
// ============================================================
enum { EP_NONE = 0, EP_PHI = 1, EP_SILU_BF16 = 2, EP_MUL_BF16 = 3, EP_ADD_RES = 4 };
constexpr int BM = 128, BN = 128, BK = 64;

__global__ void __launch_bounds__(256)
gemm_bf16_nt_kernel(const unsigned short* __restrict__ A,
                    const unsigned short* __restrict__ BT,
                    void* __restrict__ Cout,
                    const unsigned short* __restrict__ Aux,   // bf16 [M,N] (EP_MUL)
                    const float* __restrict__ Res,            // f32  [M,N] (EP_ADD_RES)
                    int Ndim, int Kdim, int ep) {
  __shared__ __align__(16) unsigned short sA[2][BM * BK];
  __shared__ __align__(16) unsigned short sB[2][BN * BK];
  const int tid  = threadIdx.x;
  const int lane = tid & 31;
  const int wave = tid >> 5;
  const int lm   = lane & 15;
  const int hh   = lane >> 4;
  const int wm   = (wave >> 1) * 32;
  const int wn   = (wave & 1) * 64;
  const int m0   = blockIdx.y * BM;
  const int n0   = blockIdx.x * BN;
  const int nk   = Kdim / BK;

  v8f acc[2][4];
#pragma unroll
  for (int i = 0; i < 2; ++i)
#pragma unroll
    for (int j = 0; j < 4; ++j)
#pragma unroll
      for (int e = 0; e < 8; ++e) acc[i][j][e] = 0.f;

#if USE_TDM
  const unsigned dwK = (unsigned)(Kdim >> 1);
  if (wave == 0) {
    tdm_load_tile((unsigned)(uintptr_t)(void*)&sA[0][0],
                  A  + (size_t)m0 * Kdim, BM, BK / 2, dwK);
    tdm_load_tile((unsigned)(uintptr_t)(void*)&sB[0][0],
                  BT + (size_t)n0 * Kdim, BN, BK / 2, dwK);
  }
#endif

  for (int it = 0; it < nk; ++it) {
    const int buf = it & 1;
#if USE_TDM
    if (wave == 0) {
      if (it + 1 < nk) {
        const int k1 = (it + 1) * BK;
        tdm_load_tile((unsigned)(uintptr_t)(void*)&sA[buf ^ 1][0],
                      A  + (size_t)m0 * Kdim + k1, BM, BK / 2, dwK);
        tdm_load_tile((unsigned)(uintptr_t)(void*)&sB[buf ^ 1][0],
                      BT + (size_t)n0 * Kdim + k1, BN, BK / 2, dwK);
        __builtin_amdgcn_s_wait_tensorcnt(2);
      } else {
        __builtin_amdgcn_s_wait_tensorcnt(0);
      }
    }
    __syncthreads();
#else
    {
      const int k0 = it * BK;
      v4u ra[4], rb[4];
#pragma unroll
      for (int p = 0; p < 4; ++p) {
        int idx = p * 2048 + tid * 8;
        int r = idx >> 6, c = idx & 63;
        ra[p] = *(const v4u*)(A  + (size_t)(m0 + r) * Kdim + k0 + c);
        rb[p] = *(const v4u*)(BT + (size_t)(n0 + r) * Kdim + k0 + c);
      }
      __syncthreads();
#pragma unroll
      for (int p = 0; p < 4; ++p) {
        int idx = p * 2048 + tid * 8;
        *(v4u*)(&sA[buf][0] + idx) = ra[p];
        *(v4u*)(&sB[buf][0] + idx) = rb[p];
      }
      __syncthreads();
    }
#endif

    const unsigned short* tA = &sA[buf][0];
    const unsigned short* tB = &sB[buf][0];
#pragma unroll
    for (int kk = 0; kk < BK; kk += 32) {
      FragU aF[2], bF[4];
#pragma unroll
      for (int i = 0; i < 2; ++i) {
        int ra2 = wm + i * 16 + lm;
        aF[i].s.lo = *(const v4u*)(tA + ra2 * BK + kk + hh * 8);
        aF[i].s.hi = *(const v4u*)(tA + ra2 * BK + kk + 16 + hh * 8);
      }
#pragma unroll
      for (int j = 0; j < 4; ++j) {
        int rb2 = wn + j * 16 + lm;
        bF[j].s.lo = *(const v4u*)(tB + rb2 * BK + kk + hh * 8);
        bF[j].s.hi = *(const v4u*)(tB + rb2 * BK + kk + 16 + hh * 8);
      }
#pragma unroll
      for (int i = 0; i < 2; ++i)
#pragma unroll
        for (int j = 0; j < 4; ++j)
          acc[i][j] = __builtin_amdgcn_wmma_f32_16x16x32_bf16(
              false, aF[i].bf, false, bF[j].bf, (short)0, acc[i][j], false, false);
    }
    __syncthreads();
  }

#pragma unroll
  for (int i = 0; i < 2; ++i)
#pragma unroll
    for (int j = 0; j < 4; ++j)
#pragma unroll
      for (int r = 0; r < 8; ++r) {
        int row = m0 + wm + i * 16 + hh * 8 + r;
        int col = n0 + wn + j * 16 + lm;
        size_t idx = (size_t)row * Ndim + col;
        float a = acc[i][j][r];
        if (ep == EP_NONE) {
          ((float*)Cout)[idx] = a;
        } else if (ep == EP_PHI) {
          ((float*)Cout)[idx] = (a > 0.f) ? (a + 1.f) : __expf(a);
        } else if (ep == EP_SILU_BF16) {
          ((unsigned short*)Cout)[idx] = f2bf(a / (1.f + __expf(-a)));
        } else if (ep == EP_MUL_BF16) {
          ((unsigned short*)Cout)[idx] = f2bf(a * bf2f(Aux[idx]));
        } else {
          ((float*)Cout)[idx] = a + Res[idx];
        }
      }
}

// ============================================================
// Linear-attention einsums on the f32 matrix pipe
// (V_WMMA_F32_16X16X4_F32; fallback = VALU).
// ============================================================
#if USE_WMMA_F32

// kv over time: C[64,64] = K^T[64,T] x V[T,64] per (b,n,h); bid=(b*N+n)*H+h
__global__ void __launch_bounds__(256)
kv_time_kernel(const float* __restrict__ Kf, const float* __restrict__ Vf,
               float* __restrict__ KV, float* __restrict__ KS) {
  int bid = blockIdx.x;
  int h = bid % H_, n = (bid / H_) % N_, b = bid / (H_ * N_);
  const int tid = threadIdx.x, lane = tid & 31, wave = tid >> 5;
  const int lm = lane & 15, hh = lane >> 4;
  const int rt = wave >> 1;             // dhk row-tile 0..3
  const int c0 = (wave & 1) * 32;       // dhv col base
  __shared__ __align__(16) float sK[32][64];
  __shared__ __align__(16) float sV[32][64];
  v8f acc[2];
#pragma unroll
  for (int c = 0; c < 2; ++c)
#pragma unroll
    for (int e = 0; e < 8; ++e) acc[c][e] = 0.f;
  float ksacc = 0.f;
  const int trow = tid >> 3;            // 0..31
  const int cst  = (tid & 7) * 8;       // 0..56
  for (int tc = 0; tc < T_; tc += 32) {
    size_t m = ((size_t)(b * T_ + tc + trow) * N_ + n);
    const float4* kp = (const float4*)(Kf + m * D_ + h * 64 + cst);
    const float4* vp = (const float4*)(Vf + m * D_ + h * 64 + cst);
    float4 k0 = kp[0], k1 = kp[1], v0 = vp[0], v1 = vp[1];
    __syncthreads();
    *(float4*)&sK[trow][cst]     = k0;
    *(float4*)&sK[trow][cst + 4] = k1;
    *(float4*)&sV[trow][cst]     = v0;
    *(float4*)&sV[trow][cst + 4] = v1;
    __syncthreads();
    if (tid < 64)
      for (int t = 0; t < 32; ++t) ksacc += sK[t][tid];
#pragma unroll
    for (int t4 = 0; t4 < 8; ++t4) {
      int tb = t4 * 4 + 2 * hh;
      v2f av;  av[0]  = sK[tb][rt * 16 + lm];  av[1]  = sK[tb + 1][rt * 16 + lm];
      v2f bv0; bv0[0] = sV[tb][c0 + lm];       bv0[1] = sV[tb + 1][c0 + lm];
      v2f bv1; bv1[0] = sV[tb][c0 + 16 + lm];  bv1[1] = sV[tb + 1][c0 + 16 + lm];
      acc[0] = __builtin_amdgcn_wmma_f32_16x16x4_f32(false, av, false, bv0, (short)0, acc[0], false, false);
      acc[1] = __builtin_amdgcn_wmma_f32_16x16x4_f32(false, av, false, bv1, (short)0, acc[1], false, false);
    }
  }
  float* kvout = KV + (size_t)bid * 4096;
#pragma unroll
  for (int c = 0; c < 2; ++c)
#pragma unroll
    for (int r = 0; r < 8; ++r)
      kvout[(rt * 16 + hh * 8 + r) * 64 + c0 + c * 16 + lm] = acc[c][r];
  if (tid < 64) KS[(size_t)bid * 64 + tid] = ksacc;
}

// kv over entities: C[64,64] = K^T[64,N] x V[N,64] per (b,t,h); bid=(b*T+t)*H+h
__global__ void __launch_bounds__(256)
kv_ent_kernel(const float* __restrict__ Kf, const float* __restrict__ Vf,
              float* __restrict__ KV, float* __restrict__ KS) {
  int bid = blockIdx.x;
  int h = bid % H_, t = (bid / H_) % T_, b = bid / (H_ * T_);
  const int tid = threadIdx.x, lane = tid & 31, wave = tid >> 5;
  const int lm = lane & 15, hh = lane >> 4;
  const int rt = wave >> 1;
  const int c0 = (wave & 1) * 32;
  __shared__ __align__(16) float sK[32][64];
  __shared__ __align__(16) float sV[32][64];
  v8f acc[2];
#pragma unroll
  for (int c = 0; c < 2; ++c)
#pragma unroll
    for (int e = 0; e < 8; ++e) acc[c][e] = 0.f;
  float ksacc = 0.f;
  const int nrow = tid >> 3;
  const int cst  = (tid & 7) * 8;
  for (int nc = 0; nc < N_; nc += 32) {
    size_t m = ((size_t)(b * T_ + t) * N_ + nc + nrow);
    const float4* kp = (const float4*)(Kf + m * D_ + h * 64 + cst);
    const float4* vp = (const float4*)(Vf + m * D_ + h * 64 + cst);
    float4 k0 = kp[0], k1 = kp[1], v0 = vp[0], v1 = vp[1];
    __syncthreads();
    *(float4*)&sK[nrow][cst]     = k0;
    *(float4*)&sK[nrow][cst + 4] = k1;
    *(float4*)&sV[nrow][cst]     = v0;
    *(float4*)&sV[nrow][cst + 4] = v1;
    __syncthreads();
    if (tid < 64)
      for (int r = 0; r < 32; ++r) ksacc += sK[r][tid];
#pragma unroll
    for (int t4 = 0; t4 < 8; ++t4) {
      int tb = t4 * 4 + 2 * hh;
      v2f av;  av[0]  = sK[tb][rt * 16 + lm];  av[1]  = sK[tb + 1][rt * 16 + lm];
      v2f bv0; bv0[0] = sV[tb][c0 + lm];       bv0[1] = sV[tb + 1][c0 + lm];
      v2f bv1; bv1[0] = sV[tb][c0 + 16 + lm];  bv1[1] = sV[tb + 1][c0 + 16 + lm];
      acc[0] = __builtin_amdgcn_wmma_f32_16x16x4_f32(false, av, false, bv0, (short)0, acc[0], false, false);
      acc[1] = __builtin_amdgcn_wmma_f32_16x16x4_f32(false, av, false, bv1, (short)0, acc[1], false, false);
    }
  }
  float* kvout = KV + (size_t)bid * 4096;
#pragma unroll
  for (int c = 0; c < 2; ++c)
#pragma unroll
    for (int r = 0; r < 8; ++r)
      kvout[(rt * 16 + hh * 8 + r) * 64 + c0 + c * 16 + lm] = acc[c][r];
  if (tid < 64) KS[(size_t)bid * 64 + tid] = ksacc;
}

// out_t[256,64] = (Q[256,64] x KV[64,64]) * rowscale(1/(q.ks+eps))
__global__ void __launch_bounds__(256)
out_time_kernel(const float* __restrict__ Q, const float* __restrict__ KV,
                const float* __restrict__ KS, float* __restrict__ Out) {
  int bid = blockIdx.x;
  int h = bid % H_, n = (bid / H_) % N_, b = bid / (H_ * N_);
  const int tid = threadIdx.x, lane = tid & 31, wave = tid >> 5;
  const int lm = lane & 15, hh = lane >> 4;
  __shared__ __align__(16) float sKV[64][64];
  __shared__ __align__(16) float sQ[256][64];
  __shared__ float sks[64];
  __shared__ float sInv[256];
  const float* kvp = KV + (size_t)bid * 4096;
  {
    const float4* kv4 = (const float4*)kvp;
    float4* skv4 = (float4*)&sKV[0][0];
#pragma unroll
    for (int p = 0; p < 4; ++p) skv4[p * 256 + tid] = kv4[p * 256 + tid];
  }
  if (tid < 64) sks[tid] = KS[(size_t)bid * 64 + tid];
#pragma unroll
  for (int p = 0; p < 16; ++p) {
    int trow = p * 16 + (tid >> 4);
    int col  = (tid & 15) * 4;
    size_t m = ((size_t)(b * T_ + trow) * N_ + n);
    *(float4*)&sQ[trow][col] = *(const float4*)(Q + m * D_ + h * 64 + col);
  }
  __syncthreads();
  {
    float den = EPS_;
    for (int d = 0; d < 64; ++d) den += sQ[tid][d] * sks[d];
    sInv[tid] = 1.f / den;
  }
  __syncthreads();
  v8f acc[2][4];
#pragma unroll
  for (int i = 0; i < 2; ++i)
#pragma unroll
    for (int j = 0; j < 4; ++j)
#pragma unroll
      for (int e = 0; e < 8; ++e) acc[i][j][e] = 0.f;
  for (int k4 = 0; k4 < 16; ++k4) {
    int kb = k4 * 4 + 2 * hh;
    v2f a0, a1;
    a0[0] = sQ[wave * 32 + lm][kb];      a0[1] = sQ[wave * 32 + lm][kb + 1];
    a1[0] = sQ[wave * 32 + 16 + lm][kb]; a1[1] = sQ[wave * 32 + 16 + lm][kb + 1];
    v2f bv[4];
#pragma unroll
    for (int j = 0; j < 4; ++j) {
      bv[j][0] = sKV[kb][j * 16 + lm];
      bv[j][1] = sKV[kb + 1][j * 16 + lm];
    }
#pragma unroll
    for (int j = 0; j < 4; ++j) {
      acc[0][j] = __builtin_amdgcn_wmma_f32_16x16x4_f32(false, a0, false, bv[j], (short)0, acc[0][j], false, false);
      acc[1][j] = __builtin_amdgcn_wmma_f32_16x16x4_f32(false, a1, false, bv[j], (short)0, acc[1][j], false, false);
    }
  }
#pragma unroll
  for (int i = 0; i < 2; ++i)
#pragma unroll
    for (int j = 0; j < 4; ++j)
#pragma unroll
      for (int r = 0; r < 8; ++r) {
        int trow = wave * 32 + i * 16 + hh * 8 + r;
        size_t m = ((size_t)(b * T_ + trow) * N_ + n);
        Out[m * D_ + h * 64 + j * 16 + lm] = acc[i][j][r] * sInv[trow];
      }
}

// out_n[64,64] = (Q x KV)*rowscale; fused combine 0.5*(out_t+out_n) -> bf16
__global__ void __launch_bounds__(256)
out_ent_combine_kernel(const float* __restrict__ Q, const float* __restrict__ KV,
                       const float* __restrict__ KS, const float* __restrict__ OutT,
                       unsigned short* __restrict__ OutBf) {
  int bid = blockIdx.x;
  int h = bid % H_, t = (bid / H_) % T_, b = bid / (H_ * T_);
  const int tid = threadIdx.x, lane = tid & 31, wave = tid >> 5;
  const int lm = lane & 15, hh = lane >> 4;
  const int rt  = wave >> 1;            // n row-tile 0..3
  const int ctb = (wave & 1) * 2;       // col tile pair base
  __shared__ __align__(16) float sKV[64][64];
  __shared__ __align__(16) float sQ[64][64];
  __shared__ float sks[64];
  __shared__ float sInv[64];
  const float* kvp = KV + (size_t)bid * 4096;
  {
    const float4* kv4 = (const float4*)kvp;
    float4* skv4 = (float4*)&sKV[0][0];
#pragma unroll
    for (int p = 0; p < 4; ++p) skv4[p * 256 + tid] = kv4[p * 256 + tid];
  }
  if (tid < 64) sks[tid] = KS[(size_t)bid * 64 + tid];
  {
    int nrow = tid >> 2;
    int col  = (tid & 3) * 16;
    size_t m = ((size_t)(b * T_ + t) * N_ + nrow);
    const float4* qp = (const float4*)(Q + m * D_ + h * 64 + col);
    float4* sp = (float4*)&sQ[nrow][col];
#pragma unroll
    for (int e = 0; e < 4; ++e) sp[e] = qp[e];
  }
  __syncthreads();
  if (tid < 64) {
    float den = EPS_;
    for (int d = 0; d < 64; ++d) den += sQ[tid][d] * sks[d];
    sInv[tid] = 1.f / den;
  }
  __syncthreads();
  v8f acc[2];
#pragma unroll
  for (int c = 0; c < 2; ++c)
#pragma unroll
    for (int e = 0; e < 8; ++e) acc[c][e] = 0.f;
  for (int k4 = 0; k4 < 16; ++k4) {
    int kb = k4 * 4 + 2 * hh;
    v2f av; av[0] = sQ[rt * 16 + lm][kb]; av[1] = sQ[rt * 16 + lm][kb + 1];
    v2f b0; b0[0] = sKV[kb][ctb * 16 + lm];       b0[1] = sKV[kb + 1][ctb * 16 + lm];
    v2f b1; b1[0] = sKV[kb][(ctb + 1) * 16 + lm]; b1[1] = sKV[kb + 1][(ctb + 1) * 16 + lm];
    acc[0] = __builtin_amdgcn_wmma_f32_16x16x4_f32(false, av, false, b0, (short)0, acc[0], false, false);
    acc[1] = __builtin_amdgcn_wmma_f32_16x16x4_f32(false, av, false, b1, (short)0, acc[1], false, false);
  }
#pragma unroll
  for (int c = 0; c < 2; ++c)
#pragma unroll
    for (int r = 0; r < 8; ++r) {
      int nr = rt * 16 + hh * 8 + r;
      int dv = (ctb + c) * 16 + lm;
      size_t m = ((size_t)(b * T_ + t) * N_ + nr);
      float comb = 0.5f * (OutT[m * D_ + h * 64 + dv] + acc[c][r] * sInv[nr]);
      OutBf[m * D_ + h * 64 + dv] = f2bf(comb);
    }
}

#else  // !USE_WMMA_F32 — VALU fallback einsums

__global__ void __launch_bounds__(256)
kv_time_kernel(const float* __restrict__ Kf, const float* __restrict__ Vf,
               float* __restrict__ KV, float* __restrict__ KS) {
  int bid = blockIdx.x;
  int h = bid % H_, n = (bid / H_) % N_, b = bid / (H_ * N_);
  int tid = threadIdx.x;
  int tr = tid >> 4, tc = tid & 15;
  __shared__ float kk[64], vv[64];
  float acc[4][4];
  for (int a = 0; a < 4; ++a)
    for (int c = 0; c < 4; ++c) acc[a][c] = 0.f;
  float ksacc = 0.f;
  for (int t = 0; t < T_; ++t) {
    size_t m = ((size_t)(b * T_ + t) * N_ + n);
    if (tid < 64)       kk[tid]      = Kf[m * D_ + h * 64 + tid];
    else if (tid < 128) vv[tid - 64] = Vf[m * D_ + h * 64 + (tid - 64)];
    __syncthreads();
    if (tid < 64) ksacc += kk[tid];
    for (int a = 0; a < 4; ++a)
      for (int c = 0; c < 4; ++c) acc[a][c] += kk[tr * 4 + a] * vv[tc * 4 + c];
    __syncthreads();
  }
  float* kvout = KV + (size_t)bid * 4096;
  for (int a = 0; a < 4; ++a)
    for (int c = 0; c < 4; ++c)
      kvout[(tr * 4 + a) * 64 + tc * 4 + c] = acc[a][c];
  if (tid < 64) KS[(size_t)bid * 64 + tid] = ksacc;
}

__global__ void __launch_bounds__(256)
kv_ent_kernel(const float* __restrict__ Kf, const float* __restrict__ Vf,
              float* __restrict__ KV, float* __restrict__ KS) {
  int bid = blockIdx.x;
  int h = bid % H_, t = (bid / H_) % T_, b = bid / (H_ * T_);
  int tid = threadIdx.x;
  int tr = tid >> 4, tc = tid & 15;
  __shared__ float kk[64], vv[64];
  float acc[4][4];
  for (int a = 0; a < 4; ++a)
    for (int c = 0; c < 4; ++c) acc[a][c] = 0.f;
  float ksacc = 0.f;
  for (int n = 0; n < N_; ++n) {
    size_t m = ((size_t)(b * T_ + t) * N_ + n);
    if (tid < 64)       kk[tid]      = Kf[m * D_ + h * 64 + tid];
    else if (tid < 128) vv[tid - 64] = Vf[m * D_ + h * 64 + (tid - 64)];
    __syncthreads();
    if (tid < 64) ksacc += kk[tid];
    for (int a = 0; a < 4; ++a)
      for (int c = 0; c < 4; ++c) acc[a][c] += kk[tr * 4 + a] * vv[tc * 4 + c];
    __syncthreads();
  }
  float* kvout = KV + (size_t)bid * 4096;
  for (int a = 0; a < 4; ++a)
    for (int c = 0; c < 4; ++c)
      kvout[(tr * 4 + a) * 64 + tc * 4 + c] = acc[a][c];
  if (tid < 64) KS[(size_t)bid * 64 + tid] = ksacc;
}

__global__ void __launch_bounds__(256)
out_time_kernel(const float* __restrict__ Q, const float* __restrict__ KV,
                const float* __restrict__ KS, float* __restrict__ Out) {
  int bid = blockIdx.x;
  int h = bid % H_, n = (bid / H_) % N_, b = bid / (H_ * N_);
  int tid = threadIdx.x;
  __shared__ float skv[4096];
  __shared__ float sks[64];
  const float* kvp = KV + (size_t)bid * 4096;
  for (int i = tid; i < 4096; i += 256) skv[i] = kvp[i];
  if (tid < 64) sks[tid] = KS[(size_t)bid * 64 + tid];
  __syncthreads();
  size_t m = ((size_t)(b * T_ + tid) * N_ + n);
  const float4* q4 = (const float4*)(Q + m * D_ + h * 64);
  float q[64];
  for (int e = 0; e < 16; ++e) {
    float4 x = q4[e];
    q[e * 4] = x.x; q[e * 4 + 1] = x.y; q[e * 4 + 2] = x.z; q[e * 4 + 3] = x.w;
  }
  float den = EPS_;
  for (int d = 0; d < 64; ++d) den += q[d] * sks[d];
  float inv = 1.f / den;
  for (int ch = 0; ch < 4; ++ch) {
    float o[16];
    for (int e = 0; e < 16; ++e) o[e] = 0.f;
    for (int d = 0; d < 64; ++d)
      for (int e = 0; e < 16; ++e) o[e] += q[d] * skv[d * 64 + ch * 16 + e];
    for (int e = 0; e < 16; ++e)
      Out[m * D_ + h * 64 + ch * 16 + e] = o[e] * inv;
  }
}

__global__ void __launch_bounds__(256)
out_ent_combine_kernel(const float* __restrict__ Q, const float* __restrict__ KV,
                       const float* __restrict__ KS, const float* __restrict__ OutT,
                       unsigned short* __restrict__ OutBf) {
  int bid = blockIdx.x;
  int h = bid % H_, t = (bid / H_) % T_, b = bid / (H_ * T_);
  int tid = threadIdx.x;
  int n = tid & 63, qt = tid >> 6;
  __shared__ float skv[4096];
  __shared__ float sks[64];
  const float* kvp = KV + (size_t)bid * 4096;
  for (int i = tid; i < 4096; i += 256) skv[i] = kvp[i];
  if (tid < 64) sks[tid] = KS[(size_t)bid * 64 + tid];
  __syncthreads();
  size_t m = ((size_t)(b * T_ + t) * N_ + n);
  const float4* q4 = (const float4*)(Q + m * D_ + h * 64);
  float q[64];
  for (int e = 0; e < 16; ++e) {
    float4 x = q4[e];
    q[e * 4] = x.x; q[e * 4 + 1] = x.y; q[e * 4 + 2] = x.z; q[e * 4 + 3] = x.w;
  }
  float den = EPS_;
  for (int d = 0; d < 64; ++d) den += q[d] * sks[d];
  float inv = 1.f / den;
  for (int e = 0; e < 16; ++e) {
    int dv = qt * 16 + e;
    float o = 0.f;
    for (int d = 0; d < 64; ++d) o += q[d] * skv[d * 64 + dv];
    float comb = 0.5f * (OutT[m * D_ + h * 64 + dv] + o * inv);
    OutBf[m * D_ + h * 64 + dv] = f2bf(comb);
  }
}

#endif  // USE_WMMA_F32

// ============================================================
// host-side orchestration
// ============================================================
extern "C" void kernel_launch(void* const* d_in, const int* in_sizes, int n_in,
                              void* d_out, int out_size, void* d_ws, size_t ws_size,
                              hipStream_t stream) {
  const float* X    = (const float*)d_in[0];
  const float* Z    = (const float*)d_in[1];
  const float* n1   = (const float*)d_in[2];
  const float* n2   = (const float*)d_in[3];
  const float* n3   = (const float*)d_in[4];
  const float* nz   = (const float*)d_in[5];
  const float* a_w[4] = {(const float*)d_in[6], (const float*)d_in[7],
                         (const float*)d_in[8], (const float*)d_in[9]};
  const float* c_w[4] = {(const float*)d_in[10], (const float*)d_in[11],
                         (const float*)d_in[12], (const float*)d_in[13]};
  const float* f_w1 = (const float*)d_in[14];
  const float* f_w2 = (const float*)d_in[15];
  const float* f_w3 = (const float*)d_in[16];
  float* Yout = (float*)d_out;

  char* base = (char*)d_ws;
  size_t off = 0;
  auto alloc = [&](size_t bytes) {
    void* p = base + off;
    off = (off + bytes + 255) & ~(size_t)255;
    return p;
  };
  unsigned short* bf_x   = (unsigned short*)alloc((size_t)M_ * D_ * 2);
  unsigned short* bf_kv  = (unsigned short*)alloc((size_t)M_ * D_ * 2);
  unsigned short* bf_out = (unsigned short*)alloc((size_t)M_ * D_ * 2);
  float* qb   = (float*)alloc((size_t)M_ * D_ * 4);
  float* kb   = (float*)alloc((size_t)M_ * D_ * 4);
  float* vb   = (float*)alloc((size_t)M_ * D_ * 4);
  float* kvt  = (float*)alloc((size_t)B_ * N_ * H_ * 4096 * 4);
  float* kst  = (float*)alloc((size_t)B_ * N_ * H_ * 64 * 4);
  float* kvn  = (float*)alloc((size_t)B_ * T_ * H_ * 4096 * 4);
  float* ksn  = (float*)alloc((size_t)B_ * T_ * H_ * 64 * 4);
  float* outt = (float*)alloc((size_t)M_ * D_ * 4);
  float* hbuf = (float*)alloc((size_t)M_ * D_ * 4);
  float* h2   = (float*)alloc((size_t)M_ * D_ * 4);
  unsigned short* u_bf    = (unsigned short*)alloc((size_t)M_ * DF_ * 2);
  unsigned short* prod_bf = (unsigned short*)alloc((size_t)M_ * DF_ * 2);
  unsigned short* aT[4], *cT[4];
  for (int i = 0; i < 4; ++i) aT[i] = (unsigned short*)alloc((size_t)D_ * D_ * 2);
  for (int i = 0; i < 4; ++i) cT[i] = (unsigned short*)alloc((size_t)D_ * D_ * 2);
  unsigned short* w1T = (unsigned short*)alloc((size_t)D_ * DF_ * 2);
  unsigned short* w3T = (unsigned short*)alloc((size_t)D_ * DF_ * 2);
  unsigned short* w2T = (unsigned short*)alloc((size_t)D_ * DF_ * 2);

  for (int i = 0; i < 4; ++i) {
    wT_bf16_kernel<<<(D_ * D_ + 255) / 256, 256, 0, stream>>>(a_w[i], aT[i], D_, D_);
    wT_bf16_kernel<<<(D_ * D_ + 255) / 256, 256, 0, stream>>>(c_w[i], cT[i], D_, D_);
  }
  wT_bf16_kernel<<<(D_ * DF_ + 255) / 256, 256, 0, stream>>>(f_w1, w1T, D_, DF_);
  wT_bf16_kernel<<<(D_ * DF_ + 255) / 256, 256, 0, stream>>>(f_w3, w3T, D_, DF_);
  wT_bf16_kernel<<<(D_ * DF_ + 255) / 256, 256, 0, stream>>>(f_w2, w2T, DF_, D_);

  dim3 g512(D_ / BN, M_ / BM);
  dim3 gDF(DF_ / BN, M_ / BM);
  const int nbT = B_ * N_ * H_;   // 2048
  const int nbN = B_ * T_ * H_;   // 8192

  auto lin_att = [&](const unsigned short* bq, const unsigned short* bkv,
                     unsigned short* const* wT4, const float* res, float* hout) {
    gemm_bf16_nt_kernel<<<g512, 256, 0, stream>>>(bq,  wT4[0], qb, nullptr, nullptr, D_, D_, EP_PHI);
    gemm_bf16_nt_kernel<<<g512, 256, 0, stream>>>(bkv, wT4[1], kb, nullptr, nullptr, D_, D_, EP_PHI);
    gemm_bf16_nt_kernel<<<g512, 256, 0, stream>>>(bkv, wT4[2], vb, nullptr, nullptr, D_, D_, EP_NONE);
    kv_time_kernel<<<nbT, 256, 0, stream>>>(kb, vb, kvt, kst);
    out_time_kernel<<<nbT, 256, 0, stream>>>(qb, kvt, kst, outt);
    kv_ent_kernel<<<nbN, 256, 0, stream>>>(kb, vb, kvn, ksn);
    out_ent_combine_kernel<<<nbN, 256, 0, stream>>>(qb, kvn, ksn, outt, bf_out);
    gemm_bf16_nt_kernel<<<g512, 256, 0, stream>>>(bf_out, wT4[3], hout, nullptr, res, D_, D_, EP_ADD_RES);
  };

  rmsnorm_bf16_kernel<<<M_ / 8, 256, 0, stream>>>(X, n1, bf_x, M_);
  lin_att(bf_x, bf_x, aT, X, hbuf);

  rmsnorm_bf16_kernel<<<M_ / 8, 256, 0, stream>>>(hbuf, n3, bf_x, M_);
  rmsnorm_bf16_kernel<<<M_ / 8, 256, 0, stream>>>(Z, nz, bf_kv, M_);
  lin_att(bf_x, bf_kv, cT, hbuf, h2);

  rmsnorm_bf16_kernel<<<M_ / 8, 256, 0, stream>>>(h2, n2, bf_x, M_);
  gemm_bf16_nt_kernel<<<gDF, 256, 0, stream>>>(bf_x, w1T, u_bf, nullptr, nullptr, DF_, D_, EP_SILU_BF16);
  gemm_bf16_nt_kernel<<<gDF, 256, 0, stream>>>(bf_x, w3T, prod_bf, u_bf, nullptr, DF_, D_, EP_MUL_BF16);
  gemm_bf16_nt_kernel<<<g512, 256, 0, stream>>>(prod_bf, w2T, Yout, nullptr, h2, D_, DF_, EP_ADD_RES);

  (void)in_sizes; (void)n_in; (void)out_size; (void)ws_size;
}
